// TransformerDiffusion_53180285059222
// MI455X (gfx1250) — compile-verified, matrix operand order checked by reading, not run on hardware
//
#include <hip/hip_runtime.h>
#include <math.h>

// ---------------------------------------------------------------------------
// Types / WMMA helpers (CDNA5 gfx1250, wave32, v_wmma_f32_16x16x32_f16)
// ---------------------------------------------------------------------------
typedef _Float16 h16;
typedef __attribute__((ext_vector_type(8)))  _Float16 v8h;
typedef __attribute__((ext_vector_type(16))) _Float16 v16h;
typedef __attribute__((ext_vector_type(8)))  float    v8f;

union V16H { v16h v; v8h h[2]; };

__device__ inline v8f v8f_zero() {
    v8f z;
#pragma unroll
    for (int i = 0; i < 8; ++i) z[i] = 0.f;
    return z;
}

#define USE_ASYNC_LDS 1

// Async copy of 16B per lane into LDS (gfx1250 GLOBAL_LOAD_ASYNC_TO_LDS_B128,
// tracked by ASYNCcnt). LDS address = low 32 bits of generic pointer
__device__ inline void async_copy_b128(h16* lds_dst, const h16* gsrc) {
#if USE_ASYNC_LDS
    unsigned lds = (unsigned)(unsigned long long)lds_dst;
    unsigned long long ga = (unsigned long long)gsrc;
    asm volatile("global_load_async_to_lds_b128 %0, %1, off"
                 :: "v"(lds), "v"(ga) : "memory");
#else
    *(v8h*)lds_dst = *(const v8h*)gsrc;
#endif
}
__device__ inline void async_wait() {
#if USE_ASYNC_LDS
    asm volatile("s_wait_asynccnt 0" ::: "memory");
#endif
}

// ---------------------------------------------------------------------------
// Tiled GEMM / fused conv1d kernel.
//   out[b, oc, l] = sum_t sum_ic W[t][oc][ic] * X[b, ic, l+t-PAD] (+bias etc.)
// W: f16 [KT][OC][IC]   X: f16 strided   Out: f16 or f32 strided
// Block tile 64(M=oc) x 256(N=b*l) x 32(K=ic); 8 waves (2Mx4N), wave = 32x64.
// Double-buffered LDS; A (weight) tile staged via async-to-LDS DMA; B tile
// staged transposed [N][K] so B fragments mirror the 16-bit A fragment
// K-interleave from CDNA5 ISA.
// ---------------------------------------------------------------------------
#define GBM 64
#define GBN 256
#define GBK 32
#define GSTR 40 // LDS row stride in halves (80B, 16B-aligned, bank-friendly)

template <int KT, bool ADD, bool RES, bool OUTF32>
__global__ __launch_bounds__(256) void gemm_conv_k(
    const h16* __restrict__ W, const h16* __restrict__ X,
    const float* __restrict__ bias, void* __restrict__ Out,
    const h16* __restrict__ Resid,
    int OC, int IC, int Lout, int Lin, int Ntot,
    long in_bstride, long out_bstride, long res_bstride,
    int in_Ls, int out_Ls, int res_Ls)
{
    constexpr int PAD = KT / 2;
    __shared__ h16 As[2][GBM * GSTR];
    __shared__ h16 Bs[2][GBN * GSTR];

    const int tid = threadIdx.x;
    const int n0  = blockIdx.x * GBN;
    const int oc0 = blockIdx.y * GBM;

    // B staging: one column (n) per thread, all 32 k's.
    const int  n_me   = n0 + tid;
    const bool nvalid = n_me < Ntot;
    const int  b_me   = nvalid ? (n_me / Lout) : 0;
    const int  l_me   = n_me - b_me * Lout;
    const h16* Xb     = X + (long)b_me * in_bstride;

    // A staging: one b128 per thread (row ar, halves ac..ac+7). 256 threads
    // cover the full 64x32 weight tile.
    const int ar = tid >> 2;
    const int ac = (tid & 3) << 3;

    const int wave = tid >> 5;
    const int lane = tid & 31;
    const int wm   = wave & 1;   // 2 M-waves
    const int wn   = wave >> 1;  // 4 N-waves
    const int frow = lane & 15;
    const int fklo = (lane & 16) ? 8 : 0;

    const int nkb    = IC / GBK;
    const int nstage = KT * nkb;

    v8f c[2][4];
#pragma unroll
    for (int i = 0; i < 2; ++i)
#pragma unroll
        for (int j = 0; j < 4; ++j) c[i][j] = v8f_zero();

    auto stage = [&](int st, int buf) {
        const int t  = st / nkb;
        const int k0 = (st - t * nkb) * GBK;
        // A tile: async DMA, 16B per thread
        async_copy_b128(&As[buf][ar * GSTR + ac],
                        W + (long)t * OC * IC + (long)(oc0 + ar) * IC + (k0 + ac));
        // B tile: transposed gather [n][k] (coalesced across lanes per k)
        h16* d = &Bs[buf][tid * GSTR];
        const int li = l_me + t - PAD;
        if (nvalid && li >= 0 && li < Lin) {
            const h16* s = Xb + li + (long)k0 * in_Ls;
#pragma unroll
            for (int kk = 0; kk < GBK; ++kk) d[kk] = s[(long)kk * in_Ls];
        } else {
#pragma unroll
            for (int kk = 0; kk < GBK; ++kk) d[kk] = (h16)0.f;
        }
    };

    stage(0, 0);
    int cur = 0;
    for (int st = 0; st < nstage; ++st) {
        async_wait();        // drain this wave's async writes to LDS
        __syncthreads();     // all waves' tile data visible
        if (st + 1 < nstage) stage(st + 1, cur ^ 1); // overlap next stage

        V16H a[2], b[4];
#pragma unroll
        for (int i = 0; i < 2; ++i) {
            const h16* p = &As[cur][(wm * 32 + i * 16 + frow) * GSTR + fklo];
            a[i].h[0] = *(const v8h*)p;
            a[i].h[1] = *(const v8h*)(p + 16);
        }
#pragma unroll
        for (int j = 0; j < 4; ++j) {
            const h16* p = &Bs[cur][(wn * 64 + j * 16 + frow) * GSTR + fklo];
            b[j].h[0] = *(const v8h*)p;
            b[j].h[1] = *(const v8h*)(p + 16);
        }
#pragma unroll
        for (int i = 0; i < 2; ++i)
#pragma unroll
            for (int j = 0; j < 4; ++j)
                c[i][j] = __builtin_amdgcn_wmma_f32_16x16x32_f16(
                    false, a[i].v, false, b[j].v, (short)0, c[i][j], false, false);
        cur ^= 1;
    }

    // Epilogue: C frag layout — lane&15 = n col, VGPR g = row g (+8 hi half)
#pragma unroll
    for (int j = 0; j < 4; ++j) {
        const int n = n0 + wn * 64 + j * 16 + (lane & 15);
        if (n >= Ntot) continue;
        const int bb = n / Lout;
        const int l  = n - bb * Lout;
#pragma unroll
        for (int i = 0; i < 2; ++i) {
            const int ocb = oc0 + wm * 32 + i * 16 + ((lane & 16) ? 8 : 0);
#pragma unroll
            for (int g = 0; g < 8; ++g) {
                const int oc = ocb + g;
                float v = c[i][j][g];
                if (bias) v += bias[oc];
                if (RES) v += (float)Resid[(long)bb * res_bstride + (long)oc * res_Ls + l];
                const long off = (long)bb * out_bstride + (long)oc * out_Ls + l;
                if (OUTF32) {
                    float* o = (float*)Out;
                    if (ADD) v += o[off];
                    o[off] = v;
                } else {
                    h16* o = (h16*)Out;
                    if (ADD) v += (float)o[off];
                    o[off] = (h16)v;
                }
            }
        }
    }
}

// ---------------------------------------------------------------------------
// Flash-style local attention. One wave per (b, head, 16-row query tile).
// qkv: f16 [B][768][LA]  (per head: q at h*192, k at h*192+64, v at h*192+128)
// abuf: f16 [B][256][LA]  (heads concatenated)
// Scores scaled by 1/8 (= (64^-0.25)^2); local mask |i-j|<64 | i<8 | j<8.
// ---------------------------------------------------------------------------
#define LA 416
#define QSTR 72
#define KSTR 72
#define PSTR 40

__global__ __launch_bounds__(32) void attn_k(const h16* __restrict__ qkv,
                                             h16* __restrict__ abuf)
{
    __shared__ h16 Qs[16 * QSTR];
    __shared__ h16 Ks[32 * KSTR];
    __shared__ h16 Ps[16 * PSTR];

    const int blk = blockIdx.x;
    const int qt  = blk % 26;
    const int h   = (blk / 26) & 3;
    const int b   = blk / (26 * 4);
    const int lane = threadIdx.x;

    const long qbase = ((long)b * 768 + h * 192) * LA;
    const h16* qp = qkv + qbase;
    const h16* kp = qkv + qbase + (long)64 * LA;
    const h16* vp = qkv + qbase + (long)128 * LA;

    // stage Q tile transposed: Qs[t][c]
    for (int idx = lane; idx < 16 * 64; idx += 32) {
        int t = idx & 15, cc = idx >> 4;
        Qs[t * QSTR + cc] = qp[(long)cc * LA + qt * 16 + t];
    }
    __syncthreads();

    const int frow = lane & 15;
    const int fklo = (lane & 16) ? 8 : 0;

    float mrow[8], lrow[8];
    v8f acc[4];
#pragma unroll
    for (int ct = 0; ct < 4; ++ct) acc[ct] = v8f_zero();
#pragma unroll
    for (int g = 0; g < 8; ++g) { mrow[g] = -1e30f; lrow[g] = 0.f; }

    V16H aq0, aq1;
    {
        const h16* p = Qs + frow * QSTR + fklo;
        aq0.h[0] = *(const v8h*)p;        aq0.h[1] = *(const v8h*)(p + 16);
        aq1.h[0] = *(const v8h*)(p + 32); aq1.h[1] = *(const v8h*)(p + 48);
    }

    const int q_lo = qt * 16;
    for (int s0 = 0; s0 < LA; s0 += 32) {
        // local-window skip: keep block 0 (j<8 region) and |i-j|<64 band
        if (!((s0 < 32) || ((s0 + 31 >= q_lo - 63) && (s0 <= q_lo + 15 + 63)))) continue;

        // stage K tile transposed: Ks[s][c]
        for (int idx = lane; idx < 32 * 64; idx += 32) {
            int sloc = idx & 31, cc = idx >> 5;
            Ks[sloc * KSTR + cc] = kp[(long)cc * LA + s0 + sloc];
        }
        __syncthreads();

        v8f sc[2]; sc[0] = v8f_zero(); sc[1] = v8f_zero();
#pragma unroll
        for (int nt = 0; nt < 2; ++nt) {
            const h16* p = Ks + (nt * 16 + frow) * KSTR + fklo;
            V16H bk0, bk1;
            bk0.h[0] = *(const v8h*)p;        bk0.h[1] = *(const v8h*)(p + 16);
            bk1.h[0] = *(const v8h*)(p + 32); bk1.h[1] = *(const v8h*)(p + 48);
            sc[nt] = __builtin_amdgcn_wmma_f32_16x16x32_f16(false, aq0.v, false, bk0.v, (short)0, sc[nt], false, false);
            sc[nt] = __builtin_amdgcn_wmma_f32_16x16x32_f16(false, aq1.v, false, bk1.v, (short)0, sc[nt], false, false);
        }
        __syncthreads();

        // online softmax; row reduction across the 16-lane half via shfl_xor
#pragma unroll
        for (int g = 0; g < 8; ++g) {
            const int m    = g + ((lane & 16) ? 8 : 0);
            const int iabs = q_lo + m;
            const int j0   = s0 + (lane & 15);
            const int j1   = j0 + 16;
            float v0 = sc[0][g] * 0.125f;
            float v1 = sc[1][g] * 0.125f;
            bool ok0 = (iabs < 392) && (j0 < 392) &&
                       (((iabs - j0 < 64) && (j0 - iabs < 64)) || iabs < 8 || j0 < 8);
            bool ok1 = (iabs < 392) && (j1 < 392) &&
                       (((iabs - j1 < 64) && (j1 - iabs < 64)) || iabs < 8 || j1 < 8);
            v0 = ok0 ? v0 : -1e30f;
            v1 = ok1 ? v1 : -1e30f;

            float rmax = fmaxf(v0, v1);
#pragma unroll
            for (int xm = 1; xm < 16; xm <<= 1) rmax = fmaxf(rmax, __shfl_xor(rmax, xm, 32));
            float mnew = fmaxf(mrow[g], rmax);
            float p0 = __expf(v0 - mnew);
            float p1 = __expf(v1 - mnew);
            float rsum = p0 + p1;
#pragma unroll
            for (int xm = 1; xm < 16; xm <<= 1) rsum += __shfl_xor(rsum, xm, 32);
            float alpha = __expf(mrow[g] - mnew);
            lrow[g] = lrow[g] * alpha + rsum;
            mrow[g] = mnew;
#pragma unroll
            for (int ct = 0; ct < 4; ++ct) acc[ct][g] *= alpha;
            Ps[m * PSTR + (lane & 15)]      = (h16)p0;
            Ps[m * PSTR + 16 + (lane & 15)] = (h16)p1;
        }
        __syncthreads();

        // PV: A frag from Ps, B frags straight from global v rows
        V16H ap;
        {
            const h16* p = Ps + frow * PSTR + fklo;
            ap.h[0] = *(const v8h*)p; ap.h[1] = *(const v8h*)(p + 16);
        }
#pragma unroll
        for (int ct = 0; ct < 4; ++ct) {
            V16H bv;
            const h16* pv = vp + (long)(ct * 16 + frow) * LA + s0 + fklo;
            bv.h[0] = *(const v8h*)pv; bv.h[1] = *(const v8h*)(pv + 16);
            acc[ct] = __builtin_amdgcn_wmma_f32_16x16x32_f16(false, ap.v, false, bv.v, (short)0, acc[ct], false, false);
        }
        __syncthreads();
    }

    h16* ob = abuf + ((long)b * 256 + h * 64) * LA + qt * 16;
#pragma unroll
    for (int ct = 0; ct < 4; ++ct) {
        const int cc = ct * 16 + (lane & 15);
#pragma unroll
        for (int g = 0; g < 8; ++g) {
            const int t = g + ((lane & 16) ? 8 : 0);
            float denom = fmaxf(lrow[g], 1e-20f);
            ob[(long)cc * LA + t] = (h16)(acc[ct][g] / denom);
        }
    }
}

// ---------------------------------------------------------------------------
// GroupNorm (+optional exact GELU / SiLU), strided f16 in/out, pad-zero tail.
// One block per (batch, group).
// ---------------------------------------------------------------------------
__global__ __launch_bounds__(256) void groupnorm_k(
    const h16* __restrict__ in, h16* __restrict__ out,
    const float* __restrict__ gamma, const float* __restrict__ beta,
    int groups, int cg, int Lvalid, int Lzero,
    long in_bstride, long out_bstride, int in_Ls, int out_Ls, int act)
{
    const int b = blockIdx.x / groups;
    const int g = blockIdx.x % groups;
    const h16* ip = in + (long)b * in_bstride + (long)g * cg * in_Ls;
    h16* op = out + (long)b * out_bstride + (long)g * cg * out_Ls;
    const int tid = threadIdx.x;

    float s = 0.f, s2 = 0.f;
    for (int c = 0; c < cg; ++c) {
        const h16* row = ip + (long)c * in_Ls;
        for (int l = tid; l < Lvalid; l += 256) {
            float v = (float)row[l];
            s += v; s2 += v * v;
        }
    }
    __shared__ float rs[256], rs2[256];
    rs[tid] = s; rs2[tid] = s2;
    __syncthreads();
    for (int o = 128; o > 0; o >>= 1) {
        if (tid < o) { rs[tid] += rs[tid + o]; rs2[tid] += rs2[tid + o]; }
        __syncthreads();
    }
    const float inv_n = 1.f / (float)(cg * Lvalid);
    const float mean  = rs[0] * inv_n;
    const float var   = rs2[0] * inv_n - mean * mean;
    const float rstd  = rsqrtf(var + 1e-5f);

    for (int c = 0; c < cg; ++c) {
        const float ga = gamma[g * cg + c], be = beta[g * cg + c];
        const h16* row = ip + (long)c * in_Ls;
        h16* orow = op + (long)c * out_Ls;
        for (int l = tid; l < Lvalid; l += 256) {
            float v = ((float)row[l] - mean) * rstd * ga + be;
            if (act == 1)      v = 0.5f * v * (1.f + erff(v * 0.70710678118f));
            else if (act == 2) v = v / (1.f + __expf(-v));
            orow[l] = (h16)v;
        }
        for (int l = Lvalid + tid; l < Lzero; l += 256) orow[l] = (h16)0.f;
    }
}

// ---------------------------------------------------------------------------
// Weight convert f32 [OC][IC][KT] -> f16 [KT][OC][IC]
// ---------------------------------------------------------------------------
__global__ __launch_bounds__(256) void cvt_w_k(const float* __restrict__ src,
                                               h16* __restrict__ dst,
                                               int OC, int IC, int KT)
{
    long i = (long)blockIdx.x * 256 + threadIdx.x;
    long tot = (long)OC * IC * KT;
    if (i >= tot) return;
    int kt = (int)(i % KT);
    long r = i / KT;
    int ic = (int)(r % IC);
    int oc = (int)(r / IC);
    dst[((long)kt * OC + oc) * IC + ic] = (h16)src[i];
}

// Strided f16 copy with zero-fill past source length.
__global__ __launch_bounds__(256) void copy_pad_k(
    const h16* __restrict__ src, h16* __restrict__ dst,
    int B, int C, int dstL, int srcL,
    long sbs, long dbs, int sLs, int dLs)
{
    long i = (long)blockIdx.x * 256 + threadIdx.x;
    long tot = (long)B * C * dstL;
    if (i >= tot) return;
    int l = (int)(i % dstL);
    long r = i / dstL;
    int c = (int)(r % C);
    int b = (int)(r / C);
    h16 v = (l < srcL) ? src[(long)b * sbs + (long)c * sLs + l] : (h16)0.f;
    dst[(long)b * dbs + (long)c * dLs + l] = v;
}

// x f32 [16][256][384] -> xin channels 0..255 (f16, bstride 1280*384)
__global__ __launch_bounds__(256) void cvt_x_k(const float* __restrict__ x,
                                               h16* __restrict__ xin)
{
    long i = (long)blockIdx.x * 256 + threadIdx.x;
    if (i >= (long)16 * 256 * 384) return;
    int l = (int)(i % 384);
    long r = i / 384;
    int c = (int)(r % 256);
    int b = (int)(r / 256);
    xin[((long)b * 1280 + c) * 384 + l] = (h16)x[i];
}

// linear-interp x_prior [16][1024][96] -> xin channels 256..1279 (f16)
__global__ __launch_bounds__(256) void interp_k(const float* __restrict__ xp,
                                                h16* __restrict__ xin)
{
    long i = (long)blockIdx.x * 256 + threadIdx.x;
    if (i >= (long)16 * 1024 * 384) return;
    int l = (int)(i % 384);
    long r = i / 384;
    int c = (int)(r % 1024);
    int b = (int)(r / 1024);
    float pos = (float)l * (95.f / 383.f);
    int i0 = (int)pos;
    int i1 = i0 + 1; if (i1 > 95) i1 = 95;
    float fr = pos - (float)i0;
    const float* row = xp + ((long)b * 1024 + c) * 96;
    float v = row[i0] * (1.f - fr) + row[i1] * fr;
    xin[((long)b * 1280 + 256 + c) * 384 + l] = (h16)v;
}

// ---------------------------------------------------------------------------
// Timestep embeddings + 2-layer MLPs + res/uncond -> blk_emb f16 [16][1024][8]
// ---------------------------------------------------------------------------
__global__ __launch_bounds__(256) void emb_k(
    const int* __restrict__ timesteps, const int* __restrict__ prior_ts,
    const int* __restrict__ resolution,
    const float* __restrict__ t_b1, const float* __restrict__ t_b2,
    const float* __restrict__ t_w1, const float* __restrict__ t_w2,
    const float* __restrict__ p_b1, const float* __restrict__ p_b2,
    const float* __restrict__ p_w1, const float* __restrict__ p_w2,
    const float* __restrict__ res_emb, const float* __restrict__ uncond,
    h16* __restrict__ blk_emb)
{
    __shared__ float emb[256];
    __shared__ float hh[256];
    const int b = blockIdx.x;
    const int tid = threadIdx.x;

    for (int pass = 0; pass < 2; ++pass) {
        float t = (float)(pass ? prior_ts[b] : timesteps[b]);
        const float* w1 = pass ? p_w1 : t_w1;
        const float* w2 = pass ? p_w2 : t_w2;
        const float* b1 = pass ? p_b1 : t_b1;
        const float* b2 = pass ? p_b2 : t_b2;
        {
            int j = tid;
            float f = (j < 128) ? __expf(-9.210340372f * (float)j / 128.f)
                                : __expf(-9.210340372f * (float)(j - 128) / 128.f);
            emb[j] = (j < 128) ? cosf(t * f) : sinf(t * f);
        }
        __syncthreads();
        {
            float acc = b1[tid];
            for (int k = 0; k < 256; ++k) acc += w1[tid * 256 + k] * emb[k];
            hh[tid] = acc / (1.f + __expf(-acc)); // silu
        }
        __syncthreads();
        for (int rr = tid; rr < 1024; rr += 256) {
            float acc = b2[rr];
            for (int k = 0; k < 256; ++k) acc += w2[rr * 256 + k] * hh[k];
            blk_emb[((long)b * 1024 + rr) * 8 + pass] = (h16)acc;
        }
        __syncthreads();
    }
    int rix = resolution[b];
    for (int rr = tid; rr < 1024; rr += 256) {
        blk_emb[((long)b * 1024 + rr) * 8 + 2] = (h16)res_emb[rix * 1024 + rr];
        for (int u = 0; u < 5; ++u)
            blk_emb[((long)b * 1024 + rr) * 8 + 3 + u] = (h16)uncond[rr * 5 + u];
    }
}

// ---------------------------------------------------------------------------
// Host orchestration
// ---------------------------------------------------------------------------
extern "C" void kernel_launch(void* const* d_in, const int* in_sizes, int n_in,
                              void* d_out, int out_size, void* d_ws, size_t ws_size,
                              hipStream_t stream)
{
    (void)in_sizes; (void)out_size; (void)ws_size;
    const int LX = 384, LAc = LA;

    // ---- params: jax tree-flatten order (nested dicts sorted by key) ----
    bool split = (n_in != 7);
    const float* blob = split ? nullptr : (const float*)d_in[2];
    size_t poff = 0; int pidx = 2;
    auto take = [&](size_t n) -> const float* {
        if (split) return (const float*)d_in[pidx++];
        const float* r = blob + poff; poff += n; return r;
    };

    const float* inp_b = take(1024);
    const float* inp_w = take((size_t)1024 * 1280 * 3);

    struct SubW {
        const float *an_b, *an_g, *norm_b, *norm_g, *proj_b, *proj_w,
                    *qkv_b, *qkv_w, *xproj_b, *xproj_w, *blk_b, *blk_w,
                    *ff_b, *ff_w, *fn_b, *fn_g;
        int IC;
    };
    struct LayerW { SubW b1, b2; const float *out_w, *pre_b, *pre_g; } Lw[8];

    auto takeSub = [&](int IC) {
        SubW s; s.IC = IC;
        s.an_b = take(256);  s.an_g = take(256);
        s.norm_b = take(IC); s.norm_g = take(IC);
        s.proj_b = take(256); s.proj_w = take((size_t)256 * 256);
        s.qkv_b = take(768);  s.qkv_w = take((size_t)768 * IC);
        s.xproj_b = take(256); s.xproj_w = take((size_t)256 * IC);
        s.blk_b = take(IC);   s.blk_w = take((size_t)IC * 1024);
        s.ff_b = take(256);   s.ff_w = take((size_t)256 * (IC + 256) * 3);
        s.fn_b = take(256);   s.fn_g = take(256);
        return s;
    };
    for (int i = 0; i < 8; ++i) {
        Lw[i].b1 = takeSub(1024);
        Lw[i].b2 = takeSub(1536);
        Lw[i].out_w = take((size_t)1024 * 1024);
        Lw[i].pre_b = take(1024);
        Lw[i].pre_g = take(1024);
    }
    const float* out_b  = take(512);
    const float* out_nb = take(1024);
    const float* out_ng = take(1024);
    const float* out_w  = take((size_t)512 * 1024 * 3);
    const float* ptime_b1 = take(256);  const float* ptime_b2 = take(1024);
    const float* ptime_w1 = take((size_t)256 * 256);
    const float* ptime_w2 = take((size_t)1024 * 256);
    const float* res_emb  = take((size_t)8 * 1024);
    const float* time_b1 = take(256);   const float* time_b2 = take(1024);
    const float* time_w1 = take((size_t)256 * 256);
    const float* time_w2 = take((size_t)1024 * 256);
    const float* uncond  = take((size_t)1024 * 5);

    const float* x       = (const float*)d_in[0];
    const float* x_prior = (const float*)d_in[1];
    const int* timesteps  = (const int*)d_in[split ? n_in - 4 : 3];
    const int* prior_ts   = (const int*)d_in[split ? n_in - 3 : 4];
    const int* resolution = (const int*)d_in[split ? n_in - 2 : 5];

    // ---- workspace carve-up (all f16, 256B aligned) ----
    char* wsb = (char*)d_ws;
    size_t wo = 0;
    auto alloc_h = [&](size_t elems) -> h16* {
        wo = (wo + 255) & ~(size_t)255;
        h16* r = (h16*)(wsb + wo);
        wo += elems * sizeof(h16);
        return r;
    };
    h16* h0   = alloc_h((size_t)16 * 1024 * LX);
    h16* h1b  = alloc_h((size_t)16 * 1024 * LX);
    h16* gno  = alloc_h((size_t)16 * 1024 * LX);
    h16* xin  = alloc_h((size_t)16 * 1280 * LX);
    h16* blkE = alloc_h((size_t)16 * 1024 * 8);
    h16* xcat = alloc_h((size_t)16 * 1536 * LAc);
    h16* gnat = alloc_h((size_t)16 * 1536 * LAc);
    h16* qkvb = alloc_h((size_t)16 * 768 * LAc);
    h16* abuf = alloc_h((size_t)16 * 256 * LAc);
    h16* aout = alloc_h((size_t)16 * 256 * LAc);
    h16* sub1 = alloc_h((size_t)16 * 1536 * LX);
    h16* sub2 = alloc_h((size_t)16 * 2048 * LX);
    h16* ffb  = alloc_h((size_t)16 * 256 * LX);
    h16* wscr = alloc_h((size_t)4 * 1024 * 1024); // per-op weight staging

    // ---- launchers ----
    auto cvt = [&](const float* src, h16* dst, int OC, int IC, int KT) {
        long tot = (long)OC * IC * KT;
        cvt_w_k<<<dim3((unsigned)((tot + 255) / 256)), dim3(256), 0, stream>>>(src, dst, OC, IC, KT);
    };
    // mode: 0 plain, 1 accumulate-into-out, 2 residual-add, 3 f32 output
    auto gemm = [&](int KT, int mode, const h16* W, const h16* X, const float* bias,
                    void* Out, const h16* Res, int OC, int IC, int Lout, int Lin,
                    long ibs, long obs, long rbs, int iLs, int oLs, int rLs) {
        const int Ntot = 16 * Lout;
        dim3 grid((unsigned)((Ntot + GBN - 1) / GBN), (unsigned)(OC / GBM));
        if (KT == 1) {
            if (mode == 1)
                gemm_conv_k<1, true, false, false><<<grid, 256, 0, stream>>>(W, X, bias, Out, Res, OC, IC, Lout, Lin, Ntot, ibs, obs, rbs, iLs, oLs, rLs);
            else if (mode == 2)
                gemm_conv_k<1, false, true, false><<<grid, 256, 0, stream>>>(W, X, bias, Out, Res, OC, IC, Lout, Lin, Ntot, ibs, obs, rbs, iLs, oLs, rLs);
            else
                gemm_conv_k<1, false, false, false><<<grid, 256, 0, stream>>>(W, X, bias, Out, Res, OC, IC, Lout, Lin, Ntot, ibs, obs, rbs, iLs, oLs, rLs);
        } else {
            if (mode == 3)
                gemm_conv_k<3, false, false, true><<<grid, 256, 0, stream>>>(W, X, bias, Out, Res, OC, IC, Lout, Lin, Ntot, ibs, obs, rbs, iLs, oLs, rLs);
            else
                gemm_conv_k<3, false, false, false><<<grid, 256, 0, stream>>>(W, X, bias, Out, Res, OC, IC, Lout, Lin, Ntot, ibs, obs, rbs, iLs, oLs, rLs);
        }
    };
    auto gnorm = [&](const h16* in, h16* out, const float* ga, const float* be,
                     int groups, int cg, int Lvalid, int Lzero,
                     long ibs, long obs, int iLs, int oLs, int act) {
        groupnorm_k<<<dim3((unsigned)(16 * groups)), 256, 0, stream>>>(
            in, out, ga, be, groups, cg, Lvalid, Lzero, ibs, obs, iLs, oLs, act);
    };
    auto copyp = [&](const h16* src, h16* dst, int C, int dstL, int srcL,
                     long sbs, long dbs, int sLs, int dLs) {
        long tot = (long)16 * C * dstL;
        copy_pad_k<<<dim3((unsigned)((tot + 255) / 256)), 256, 0, stream>>>(
            src, dst, 16, C, dstL, srcL, sbs, dbs, sLs, dLs);
    };

    // ---- embeddings / input prep ----
    emb_k<<<16, 256, 0, stream>>>(timesteps, prior_ts, resolution,
                                  time_b1, time_b2, time_w1, time_w2,
                                  ptime_b1, ptime_b2, ptime_w1, ptime_w2,
                                  res_emb, uncond, blkE);
    {
        long tot = (long)16 * 256 * 384;
        cvt_x_k<<<dim3((unsigned)((tot + 255) / 256)), 256, 0, stream>>>(x, xin);
        tot = (long)16 * 1024 * 384;
        interp_k<<<dim3((unsigned)((tot + 255) / 256)), 256, 0, stream>>>(x_prior, xin);
    }

    // ---- input conv (k=3, 1280 -> 1024) ----
    cvt(inp_w, wscr, 1024, 1280, 3);
    gemm(3, 0, wscr, xin, inp_b, h0, nullptr, 1024, 1280, LX, LX,
         (long)1280 * LX, (long)1024 * LX, 0, LX, LX, 0);

    // ---- sub_block runner ----
    auto run_sub = [&](const SubW& S, const h16* xbuf, h16* subout) {
        const int IC = S.IC;
        const long xbs  = (long)IC * LX;
        const long sobs = (long)(IC + 512) * LX;
        const long xcbs = (long)IC * LAc;

        // blk_enc -> xcat[:, :, 0:8]
        cvt(S.blk_w, wscr, IC, 1024, 1);
        gemm(1, 0, wscr, blkE, S.blk_b, xcat, nullptr, IC, 1024, 8, 8,
             (long)1024 * 8, xcbs, 0, 8, LAc, 0);
        // x -> xcat[:, :, 8:392], zero [392:416)
        copyp(xbuf, xcat + 8, IC, 408, LX, xbs, xcbs, LX, LAc);
        // attention pre-norm (32 groups over valid 392)
        gnorm(xcat, gnat, S.norm_g, S.norm_b, 32, IC / 32, 392, LAc,
              xcbs, xcbs, LAc, LAc, 0);
        // qkv (k=1, IC -> 768)
        cvt(S.qkv_w, wscr, 768, IC, 1);
        gemm(1, 0, wscr, gnat, S.qkv_b, qkvb, nullptr, 768, IC, LAc, LAc,
             xcbs, (long)768 * LAc, 0, LAc, LAc, 0);
        // xproj on (pre-norm) xcat -> aout
        cvt(S.xproj_w, wscr, 256, IC, 1);
        gemm(1, 0, wscr, xcat, S.xproj_b, aout, nullptr, 256, IC, LAc, LAc,
             xcbs, (long)256 * LAc, 0, LAc, LAc, 0);
        // masked local attention -> abuf
        attn_k<<<dim3((unsigned)(16 * 4 * 26)), 32, 0, stream>>>(qkvb, abuf);
        // proj(a) accumulated into aout
        cvt(S.proj_w, wscr, 256, 256, 1);
        gemm(1, 1, wscr, abuf, S.proj_b, aout, nullptr, 256, 256, LAc, LAc,
             (long)256 * LAc, (long)256 * LAc, 0, LAc, LAc, 0);
        // ah = aout[:, :, 8:392] -> GN(8) + GELU -> subout ch[0:256)
        gnorm(aout + 8, subout, S.an_g, S.an_b, 8, 32, LX, LX,
              (long)256 * LAc, sobs, LAc, LX, 1);
        // x -> subout ch[256:256+IC)
        copyp(xbuf, subout + (long)256 * LX, IC, LX, LX, xbs, sobs, LX, LX);
        // ff conv (k=3, IC+256 -> 256)
        cvt(S.ff_w, wscr, 256, IC + 256, 3);
        gemm(3, 0, wscr, subout, S.ff_b, ffb, nullptr, 256, IC + 256, LX, LX,
             sobs, (long)256 * LX, 0, LX, LX, 0);
        // GN(8) + GELU -> subout ch[IC+256 : IC+512)
        gnorm(ffb, subout + (long)(IC + 256) * LX, S.fn_g, S.fn_b, 8, 32, LX, LX,
              (long)256 * LX, sobs, LX, LX, 1);
    };

    // ---- layers ----
    h16* hin = h0;
    h16* hout = h1b;
    for (int i = 0; i < 8; ++i) {
        gnorm(hin, gno, Lw[i].pre_g, Lw[i].pre_b, 8, 128, LX, LX,
              (long)1024 * LX, (long)1024 * LX, LX, LX, 0);
        run_sub(Lw[i].b1, gno, sub1);
        run_sub(Lw[i].b2, sub1, sub2);
        // out conv (k=1, sub2 ch[1024:2048) -> 1024) + residual hin
        cvt(Lw[i].out_w, wscr, 1024, 1024, 1);
        gemm(1, 2, wscr, sub2 + (long)1024 * LX, nullptr, hout, hin,
             1024, 1024, LX, LX, (long)2048 * LX, (long)1024 * LX,
             (long)1024 * LX, LX, LX, LX);
        h16* t = hin; hin = hout; hout = t;
    }

    // ---- final GN(32)+SiLU then output conv (k=3, 1024 -> 512, f32 out) ----
    gnorm(hin, gno, out_ng, out_nb, 32, 32, LX, LX,
          (long)1024 * LX, (long)1024 * LX, LX, LX, 2);
    cvt(out_w, wscr, 512, 1024, 3);
    gemm(3, 3, wscr, gno, out_b, d_out, nullptr, 512, 1024, LX, LX,
         (long)1024 * LX, (long)512 * LX, 0, LX, LX, 0);
}
